// DCNModule_23527830847721
// MI455X (gfx1250) — matrix operand order repaired and checked by reading, not compiled, not run on hardware
//
#include <hip/hip_runtime.h>

// ---------------------------------------------------------------------------
// x [4,256,64,64] f32. Offset conv 3x3 pad1 -> 27ch (WMMA implicit GEMM,
// N padded to 32). Main DCN: fused bilinear sample + bf16 WMMA GEMM
// M=16384, N=256, K=2304, + bias + ReLU.  Each DCN block covers ALL 256 N
// so the expensive gather/sample pass runs exactly once per M-tile.
// ---------------------------------------------------------------------------
#define KKT    9
#define C_IN   256
#define C_OUT  256
#define Hh     64
#define Ww     64
#define HWp    4096
#define BN     4
#define OFFCH  27
#define KDIM   2304
#define NKT    (KDIM / 32)     // 72 K-steps

#define MT 128                 // M tile (pixels) per workgroup
// A LDS fragment layout: subtile s (16 rows) x lane(32) x 16 bf16, per-lane
// stride 24 bf16 (48B, 16B-aligned for b128). One buffer = 8*32*12 u32 = 12KB.
#define ABUF_U32 3072

typedef __attribute__((ext_vector_type(16))) __bf16 v16bf;
typedef __attribute__((ext_vector_type(2)))  __bf16 v2bf;
typedef __attribute__((ext_vector_type(8)))  float  v8f;
typedef __attribute__((ext_vector_type(8)))  unsigned int v8u;

static __device__ __forceinline__ unsigned short f2bf_raw(float f) {
  unsigned u = __float_as_uint(f);
  unsigned r = u + 0x7FFFu + ((u >> 16) & 1u);
  return (unsigned short)(r >> 16);
}
// packed fp32x2 -> bf16x2 (native v_cvt_pk_bf16_f32 when available)
static __device__ __forceinline__ unsigned pack_bf16(float a, float b) {
  v2bf t;
  t[0] = (__bf16)a;
  t[1] = (__bf16)b;
  return __builtin_bit_cast(unsigned, t);
}
static __device__ __forceinline__ unsigned short f2h_raw(float f) {
  return __builtin_bit_cast(unsigned short, (_Float16)f);
}
static __device__ __forceinline__ float h2f(unsigned short s) {
  return (float)__builtin_bit_cast(_Float16, s);
}

// fragment-layout address (contiguous k order used here, 16-bit A 16x32):
//   lane = (row&15) | (K.bit4 << 4) ; per-lane elem p = K&15
static __device__ __forceinline__ int a_u32_addr(int mi, int t) {
  int s  = mi >> 4;
  int ln = (mi & 15) | (((t >> 4) & 1) << 4);
  int p  = t & 15;                            // t even -> p even
  return (s * 32 + ln) * 12 + (p >> 1);
}

// ---------------------------------------------------------------------------
// Kernel 1: swizzle dcn weights -> fragment-layout bf16
//   wb[kt][ns:16][lane:32][p:16];  B frag: lane holds N=lane&15,
//   K = p + 16*(lane>>4) within each 32-K tile.
// ---------------------------------------------------------------------------
__global__ void __launch_bounds__(256) swz_dcn_kernel(
    const float* __restrict__ w_dcn, unsigned short* __restrict__ wb) {
  int e = blockIdx.x * 256 + threadIdx.x;
  if (e >= NKT * 16 * 32 * 16) return;
  int p    = e & 15;
  int lane = (e >> 4) & 31;
  int ns   = (e >> 9) & 15;
  int kt   = e >> 13;
  int n  = ns * 16 + (lane & 15);
  int k  = kt * 32 + p + ((lane >> 4) << 4);
  wb[e] = f2bf_raw(w_dcn[(long)n * KDIM + k]);
}

// Kernel 2: swizzle offset-conv weights (27 -> pad 32) -> wboff[kt][ns:2][32][16]
__global__ void __launch_bounds__(256) swz_off_kernel(
    const float* __restrict__ w_off, unsigned short* __restrict__ wboff) {
  int e = blockIdx.x * 256 + threadIdx.x;
  if (e >= NKT * 2 * 32 * 16) return;
  int p    = e & 15;
  int lane = (e >> 4) & 31;
  int ns   = (e >> 9) & 1;
  int kt   = e >> 10;
  int n  = ns * 16 + (lane & 15);
  int k  = kt * 32 + p + ((lane >> 4) << 4);
  float v = (n < OFFCH) ? w_off[(long)n * KDIM + k] : 0.f;
  wboff[e] = f2bf_raw(v);
}

// ---------------------------------------------------------------------------
// A-tile fills: branchless, one packed LDS param load per element,
// incremental (c,kk) tracking (single division per call).
// Thread -> fixed pixel mi = tid&127, t = t0 + 4i, t0 = 2*(tid>>7).
// ---------------------------------------------------------------------------
// 1-tap (im2col, offset conv): param u32 = clamped_addr | (f16 valid << 16)
static __device__ __forceinline__ void fillA1(
    unsigned int* __restrict__ buf, const float* __restrict__ xb,
    const unsigned int* __restrict__ sPk, int k0, int tid) {
  const int mi = tid & (MT - 1);
  const int t0 = ((tid >> 7) & 1) * 2;
  int k  = k0 + t0;
  int c  = k / KKT;
  int kk = k - c * KKT;
  const int pbase = mi * KKT;
#pragma unroll
  for (int i = 0; i < 8; ++i) {
    unsigned pk0 = sPk[pbase + kk];
    float v0 = h2f((unsigned short)(pk0 >> 16))
             * xb[((long)c << 12) + (pk0 & 0xFFFFu)];
    int c1 = c, kk1 = kk + 1;
    if (kk1 == KKT) { kk1 = 0; ++c1; }
    unsigned pk1 = sPk[pbase + kk1];
    float v1 = h2f((unsigned short)(pk1 >> 16))
             * xb[((long)c1 << 12) + (pk1 & 0xFFFFu)];
    buf[a_u32_addr(mi, t0 + 4 * i)] = pack_bf16(v0, v1);
    kk += 4; if (kk >= KKT) { kk -= KKT; ++c; }
  }
}

// 4-tap (bilinear, dcn): param uint4 = {i0|i1<<16, i2|i3<<16,
//                                       w0h|w1h<<16, w2h|w3h<<16}
static __device__ __forceinline__ float tap4(const float* __restrict__ xp,
                                             uint4 pk) {
  return h2f((unsigned short)(pk.z & 0xFFFFu)) * xp[pk.x & 0xFFFFu]
       + h2f((unsigned short)(pk.z >> 16))     * xp[pk.x >> 16]
       + h2f((unsigned short)(pk.w & 0xFFFFu)) * xp[pk.y & 0xFFFFu]
       + h2f((unsigned short)(pk.w >> 16))     * xp[pk.y >> 16];
}
static __device__ __forceinline__ void fillA4(
    unsigned int* __restrict__ buf, const float* __restrict__ xb,
    const uint4* __restrict__ sPk, int k0, int tid) {
  const int mi = tid & (MT - 1);
  const int t0 = ((tid >> 7) & 1) * 2;
  int k  = k0 + t0;
  int c  = k / KKT;
  int kk = k - c * KKT;
  const int pbase = mi * KKT;
#pragma unroll
  for (int i = 0; i < 8; ++i) {
    float v0 = tap4(xb + ((long)c << 12), sPk[pbase + kk]);
    int c1 = c, kk1 = kk + 1;
    if (kk1 == KKT) { kk1 = 0; ++c1; }
    float v1 = tap4(xb + ((long)c1 << 12), sPk[pbase + kk1]);
    buf[a_u32_addr(mi, t0 + 4 * i)] = pack_bf16(v0, v1);
    kk += 4; if (kk >= KKT) { kk -= KKT; ++c; }
  }
}

// ---------------------------------------------------------------------------
// Kernel 3: offset conv as WMMA implicit GEMM. M-tile 128, N=32.
// 8 waves: wave wv owns A subtile wv (16 rows) x 2 WMMAs. grid = 128 blocks.
// ---------------------------------------------------------------------------
__global__ void __launch_bounds__(256) off_gemm_kernel(
    const float* __restrict__ x, const unsigned short* __restrict__ wboff,
    const float* __restrict__ b_off, float* __restrict__ off) {
  __shared__ unsigned int sA32[2][ABUF_U32];   // 24 KB double-buffered A
  __shared__ unsigned int sPk1[MT * KKT];      // 4.6 KB im2col params
  __shared__ float sOut[32 * MT];              // 16 KB epilogue transpose

  const int tid = threadIdx.x;
  const int pixbase = blockIdx.x * MT;
  const int b = pixbase >> 12;
  const int hwbase = pixbase & (HWp - 1);
  const int lane = tid & 31;
  const int wv = tid >> 5;

  const float* xb = x + (((long)b * C_IN) << 12);

  // params: clamped address + f16 validity weight per (pixel, tap)
  for (int e = tid; e < MT * KKT; e += 256) {
    int mi = e / KKT;
    int kk = e - mi * KKT;
    int hw = hwbase + mi;
    int h = hw >> 6, w = hw & 63;
    int ky = kk / 3, kx = kk - ky * 3;
    int y = h - 1 + ky, xx = w - 1 + kx;
    bool valid = (y >= 0) && (y < Hh) && (xx >= 0) && (xx < Ww);
    int cy = min(max(y, 0), Hh - 1), cx = min(max(xx, 0), Ww - 1);
    sPk1[e] = (unsigned)(cy * Ww + cx)
            | ((unsigned)f2h_raw(valid ? 1.f : 0.f) << 16);
  }
  __syncthreads();

  v8f acc[2];
  acc[0] = {}; acc[1] = {};

  fillA1(sA32[0], xb, sPk1, 0, tid);
  __syncthreads();

  for (int kt = 0; kt < NKT; ++kt) {
    int cur = kt & 1;
    v16bf bfg[2];
#pragma unroll
    for (int in = 0; in < 2; ++in)
      bfg[in] = __builtin_bit_cast(v16bf,
          *(const v8u*)(wboff + ((((long)kt * 2 + in) * 32 + lane) << 4)));
    v16bf af = __builtin_bit_cast(v16bf,
        *(const v8u*)&sA32[cur][(wv * 32 + lane) * 12]);
    if (kt + 1 < NKT)
      fillA1(sA32[cur ^ 1], xb, sPk1, (kt + 1) * 32, tid);
#pragma unroll
    for (int in = 0; in < 2; ++in)
      acc[in] = __builtin_amdgcn_wmma_f32_16x16x32_bf16(
          false, af, false, bfg[in], (short)0, acc[in], false, false);
    __syncthreads();
  }

  // epilogue: LDS transpose -> coalesced writes of the 27 live channels
  const int nloc  = lane & 15;
  const int mbase = (lane >> 4) << 3;
#pragma unroll
  for (int in = 0; in < 2; ++in)
#pragma unroll
    for (int i = 0; i < 8; ++i)
      sOut[(in * 16 + nloc) * MT + (wv * 16 + mbase + i)] = acc[in][i];
  __syncthreads();
  for (int e = tid; e < OFFCH * MT; e += 256) {
    int m  = e & (MT - 1);
    int nl = e >> 7;
    float v = sOut[nl * MT + m] + b_off[nl];
    if (nl >= 18) v = 1.f / (1.f + __expf(-v));
    off[(((long)b * OFFCH + nl) << 12) + hwbase + m] = v;
  }
}

// ---------------------------------------------------------------------------
// Kernel 4: fused bilinear-sample + bf16 WMMA GEMM + bias + ReLU
// block 256 = 8 waves as 2(M) x 4(N); wave tile 64x64 (16 accumulators);
// block tile 128 x 256  ->  sampling runs ONCE per M-tile. grid = 128.
// ---------------------------------------------------------------------------
__global__ void __launch_bounds__(256) dcn_gemm_kernel(
    const float* __restrict__ x, const float* __restrict__ off,
    const unsigned short* __restrict__ wb, const float* __restrict__ b_dcn,
    float* __restrict__ out) {
  __shared__ __align__(16) float sScratch[64 * MT];  // 32 KB: params / sOut
  __shared__ unsigned int sA32[2][ABUF_U32];         // 24 KB double-buffered A
  uint4* sPk4 = (uint4*)sScratch;                    // 18 KB packed taps
  float* sOut = sScratch;                            // epilogue reuse

  const int tid = threadIdx.x;
  const int pixbase = blockIdx.x * MT;
  const int b       = pixbase >> 12;
  const int hwbase  = pixbase & (HWp - 1);

  // ---- precompute packed bilinear taps: {4x u16 addr, 4x f16 weight} ----
  for (int e = tid; e < MT * KKT; e += 256) {
    int mi = e / KKT;
    int kk = e - mi * KKT;
    int hw = hwbase + mi;
    int h = hw >> 6, w = hw & 63;
    const float* ob = off + (((long)b * OFFCH) << 12) + hw;
    float oy = ob[(long)kk << 12];
    float ox = ob[(long)(9 + kk) << 12];
    float mk = ob[(long)(18 + kk) << 12];
    int ky = kk / 3, kx = kk - ky * 3;
    float py = (float)(h - 1 + ky) + oy;
    float px = (float)(w - 1 + kx) + ox;
    float y0f = floorf(py), x0f = floorf(px);
    float ly = py - y0f,  lx = px - x0f;
    int y0 = (int)y0f, x0 = (int)x0f;
    int y1 = y0 + 1,   x1 = x0 + 1;
    float hy = 1.f - ly, hx = 1.f - lx;
    bool vy0 = (y0 >= 0) && (y0 < Hh), vy1 = (y1 >= 0) && (y1 < Hh);
    bool vx0 = (x0 >= 0) && (x0 < Ww), vx1 = (x1 >= 0) && (x1 < Ww);
    int cy0 = min(max(y0, 0), Hh - 1), cy1 = min(max(y1, 0), Hh - 1);
    int cx0 = min(max(x0, 0), Ww - 1), cx1 = min(max(x1, 0), Ww - 1);
    uint4 pk;
    pk.x = (unsigned)(cy0 * Ww + cx0) | ((unsigned)(cy0 * Ww + cx1) << 16);
    pk.y = (unsigned)(cy1 * Ww + cx0) | ((unsigned)(cy1 * Ww + cx1) << 16);
    pk.z = (unsigned)f2h_raw((vy0 && vx0) ? hy * hx * mk : 0.f)
         | ((unsigned)f2h_raw((vy0 && vx1) ? hy * lx * mk : 0.f) << 16);
    pk.w = (unsigned)f2h_raw((vy1 && vx0) ? ly * hx * mk : 0.f)
         | ((unsigned)f2h_raw((vy1 && vx1) ? ly * lx * mk : 0.f) << 16);
    sPk4[e] = pk;
  }
  __syncthreads();

  const int lane = tid & 31;
  const int wv = tid >> 5;
  const int wm = wv & 1;                 // M offset 64*wm
  const int wn = wv >> 1;                // N offset 64*wn

  v8f acc[4][4];
#pragma unroll
  for (int im = 0; im < 4; ++im)
#pragma unroll
    for (int in = 0; in < 4; ++in) acc[im][in] = {};

  const float* xb = x + (((long)b * C_IN) << 12);

  fillA4(sA32[0], xb, sPk4, 0, tid);
  __syncthreads();

  for (int kt = 0; kt < NKT; ++kt) {
    int cur = kt & 1;
    // A fragments from LDS (2x ds_load_b128 each)
    v16bf af[4];
#pragma unroll
    for (int im = 0; im < 4; ++im)
      af[im] = __builtin_bit_cast(v16bf,
          *(const v8u*)&sA32[cur][((wm * 4 + im) * 32 + lane) * 12]);
    if (kt + 1 < NKT) {
      __builtin_prefetch(
          (const void*)(wb +
              ((((long)(kt + 1) * 16 + wn * 4) * 32 + lane) << 4)),
          0, 1);
      fillA4(sA32[cur ^ 1], xb, sPk4, (kt + 1) * 32, tid);
    }
    // stream B fragments from L2, one at a time (2x global_load_b128 each)
#pragma unroll
    for (int in = 0; in < 4; ++in) {
      int ns = wn * 4 + in;
      v16bf bf = __builtin_bit_cast(v16bf,
          *(const v8u*)(wb + ((((long)kt * 16 + ns) * 32 + lane) << 4)));
#pragma unroll
      for (int im = 0; im < 4; ++im)
        acc[im][in] = __builtin_amdgcn_wmma_f32_16x16x32_bf16(
            false, af[im], false, bf, (short)0, acc[im][in], false, false);
    }
    __syncthreads();
  }

  // ---- epilogue: 4 chunks of 64 N-cols through LDS, coalesced NCHW writes --
  const int nloc  = lane & 15;
  const int mbase = (lane >> 4) << 3;
  for (int q = 0; q < 4; ++q) {
    if (wn == q) {
#pragma unroll
      for (int im = 0; im < 4; ++im)
#pragma unroll
        for (int in = 0; in < 4; ++in)
#pragma unroll
          for (int i = 0; i < 8; ++i)
            sOut[(in * 16 + nloc) * MT + (wm * 64 + im * 16 + mbase + i)]
                = acc[im][in][i];
    }
    __syncthreads();
    for (int e = tid; e < 64 * MT; e += 256) {
      int m  = e & (MT - 1);
      int nl = e >> 7;
      int n  = q * 64 + nl;
      float v = sOut[nl * MT + m] + b_dcn[n];
      v = v > 0.f ? v : 0.f;
      out[(((long)b * C_OUT + n) << 12) + hwbase + m] = v;
    }
    __syncthreads();
  }
}

// ---------------------------------------------------------------------------
// Host launcher.  inputs: 0:x 1:w_off 2:b_off 3:w_dcn 4:b_dcn (all f32)
// workspace: [off f32 1.77MB][wb_dcn bf16 1.18MB][wb_off bf16 0.15MB]
// ---------------------------------------------------------------------------
extern "C" void kernel_launch(void* const* d_in, const int* in_sizes, int n_in,
                              void* d_out, int out_size, void* d_ws, size_t ws_size,
                              hipStream_t stream) {
  const float* x     = (const float*)d_in[0];
  const float* w_off = (const float*)d_in[1];
  const float* b_off = (const float*)d_in[2];
  const float* w_dcn = (const float*)d_in[3];
  const float* b_dcn = (const float*)d_in[4];
  float* out = (float*)d_out;

  const size_t off_bytes = (size_t)BN * OFFCH * HWp * sizeof(float);
  const size_t wb_elems  = (size_t)NKT * 16 * 32 * 16;
  float*          d_off   = (float*)d_ws;
  unsigned short* d_wb    = (unsigned short*)((char*)d_ws + off_bytes);
  unsigned short* d_wboff = d_wb + wb_elems;

  swz_dcn_kernel<<<(int)((wb_elems + 255) / 256), 256, 0, stream>>>(w_dcn, d_wb);
  swz_off_kernel<<<(NKT * 2 * 32 * 16 + 255) / 256, 256, 0, stream>>>(w_off, d_wboff);

  off_gemm_kernel<<<(BN * HWp) / MT, 256, 0, stream>>>(x, d_wboff, b_off, d_off);

  dcn_gemm_kernel<<<(BN * HWp) / MT, 256, 0, stream>>>(x, d_off, d_wb, b_dcn, out);
}
